// EncodingLoss_35270271434961
// MI455X (gfx1250) — matrix operand order repaired and checked by reading, not compiled, not run on hardware
//
#include <hip/hip_runtime.h>
#include <hip/hip_bf16.h>

// ---------------------------------------------------------------------------
// EncodingLoss for MI455X (gfx1250, wave32, WMMA).
// Dominant work: 2048x2048x5024 f16 WMMA GEMM (cos of peak_data rows).
// GEMM: 128x128 block tile, 8 waves (wave tile 32x64 = 2x4 WMMA tiles),
// double-buffered LDS staged via GLOBAL_LOAD_ASYNC_TO_LDS_B128 when available.
// ---------------------------------------------------------------------------

typedef __attribute__((ext_vector_type(16))) _Float16 v16h;
typedef __attribute__((ext_vector_type(8)))  float    v8f;

static constexpr int   NN = 2048;   // cells
static constexpr int   DD = 64;     // embedding dim
static constexpr int   PP = 5000;   // peak feature dim
static constexpr int   PPAD = 5024; // padded to multiple of 32
static constexpr int   CC = 20;     // clusters
static constexpr float RARE_THR = 61.44f;  // N * 0.03

// ---- workspace layout (bytes) ----
static constexpr size_t PN_OFF    = 0;                                  // f16 [NN*PPAD]
static constexpr size_t GM_OFF    = PN_OFF + (size_t)NN * PPAD * 2;     // f32 [NN*NN] (pc, then ar)
static constexpr size_t ADJ_OFF   = GM_OFF + (size_t)NN * NN * 4;       // u8  [NN*NN]
static constexpr size_t ATACN_OFF = ADJ_OFF + (size_t)NN * NN;          // f16 [NN*DD]
static constexpr size_t RNAN_OFF  = ATACN_OFF + (size_t)NN * DD * 2;    // f16 [NN*DD]
static constexpr size_t FREG_OFF  = RNAN_OFF + (size_t)NN * DD * 2;     // float region

// float-region indices (in floats)
static constexpr int FI_CNTS  = 0;                 // [CC]
static constexpr int FI_S     = FI_CNTS + CC;      // [CC*DD]
static constexpr int FI_SS    = FI_S + CC * DD;    // [CC*DD]
static constexpr int FI_PS    = FI_SS + CC * DD;   // [CC*CC]
static constexpr int FI_SC    = FI_PS + CC * CC;   // [64] scalars
static constexpr int ZERO_N   = FI_SC + 64;        // everything above is zero-init'd
static constexpr int FI_MEANC = ZERO_N;            // [CC*DD]
static constexpr int FI_CMR   = FI_MEANC + CC*DD;  // [DD] col means rna
static constexpr int FI_CMA   = FI_CMR + DD;       // [DD] col means atac
static constexpr int FI_T2    = FI_CMA + DD;       // [NN]
static constexpr int FI_BSIM  = FI_T2 + NN;        // [NN]
static constexpr int FREG_FLOATS = FI_BSIM + NN;

static constexpr size_t INT_OFF   = FREG_OFF + (size_t)FREG_FLOATS * 4;
static constexpr size_t MATCH_OFF = INT_OFF + (size_t)3 * NN * 4;

// SC scalar slots
static constexpr int SC_KK = 0, SC_K2 = 1, SC_RARE = 2, SC_CSL = 3;
static constexpr int SC_SUMSTD_R = 4, SC_SUMABS_R = 5, SC_COVS_R = 6, SC_COVC_R = 7;
static constexpr int SC_SUMSTD_A = 8, SC_SUMABS_A = 9, SC_COVS_A = 10, SC_COVC_A = 11;
static constexpr int SC_NEAR = 12, SC_LMNN = 13, SC_CNT1 = 14, SC_LNEAR = 15, SC_CNT2 = 16;

// ---- CDNA5 async global->LDS staging (ASYNCcnt path), with sync fallback ----
#if defined(__has_builtin)
#if __has_builtin(__builtin_amdgcn_global_load_async_to_lds_b128) && \
    __has_builtin(__builtin_amdgcn_s_wait_asynccnt)
#define ASYNC_LDS 1
#endif
#endif
#ifndef ASYNC_LDS
#define ASYNC_LDS 0
#endif

#if ASYNC_LDS
// builtin signature (from hipcc diagnostic): params are pointers to
// 'int __attribute__((vector_size(16)))' in AS1 (global) and AS3 (LDS).
typedef int v4i_vs __attribute__((__vector_size__(16)));
typedef __attribute__((address_space(1))) v4i_vs* as1_v4i_p;
typedef __attribute__((address_space(3))) v4i_vs* as3_v4i_p;
#endif

__device__ __forceinline__ void copy16_gl_to_lds(const _Float16* g, _Float16* l) {
#if ASYNC_LDS
  __builtin_amdgcn_global_load_async_to_lds_b128((as1_v4i_p)g, (as3_v4i_p)l, 0, 0);
#else
  *(float4*)l = *(const float4*)g;
#endif
}

__device__ __forceinline__ void wait_async_copies() {
#if ASYNC_LDS
  __builtin_amdgcn_s_wait_asynccnt(0);
#endif
}

// ---------------------------------------------------------------------------
__global__ void zero_f32_kernel(float* p, int n) {
  int i = blockIdx.x * blockDim.x + threadIdx.x;
  if (i < n) p[i] = 0.f;
}

// row-normalize X[row][0..d) and emit f16 padded to kpad (zeros past d)
__global__ __launch_bounds__(256)
void rownorm_to_f16(const float* __restrict__ X, _Float16* __restrict__ Y, int d, int kpad) {
  const int row = blockIdx.x;
  const int tid = threadIdx.x;
  const float* x = X + (size_t)row * d;
  __shared__ float red[256];
  float s = 0.f;
  for (int i = tid; i < d; i += 256) { float v = x[i]; s += v * v; }
  red[tid] = s; __syncthreads();
  for (int o = 128; o > 0; o >>= 1) { if (tid < o) red[tid] += red[tid + o]; __syncthreads(); }
  const float inv = rsqrtf(red[0]);
  _Float16* y = Y + (size_t)row * kpad;
  for (int i = tid; i < kpad; i += 256)
    y[i] = (i < d) ? (_Float16)(x[i] * inv) : (_Float16)0.0f;
}

// ---------------------------------------------------------------------------
// C[M,N] = A[M,K] * B[N,K]^T. A,B f16 row-major, K % 32 == 0, M,N % 128 == 0.
// Block tile 128x128, 8 waves (4 in M, 2 in N), wave tile 32x64 = 2x4 WMMA tiles.
// LDS rows padded to 40 halfs (20 dwords) -> conflict-free 16-lane fragment reads.
#define GEMM_BM 128
#define GEMM_BN 128
#define GEMM_BK 32
#define LDSK    40   // padded row stride in halfs

__global__ __launch_bounds__(256)
void gemm_nt_f16(const _Float16* __restrict__ A, const _Float16* __restrict__ B,
                 float* __restrict__ C, int M, int N, int K) {
  __shared__ _Float16 sA[2][GEMM_BM * LDSK];
  __shared__ _Float16 sB[2][GEMM_BN * LDSK];
  const int tid  = threadIdx.x;
  const int wave = tid >> 5;
  const int lane = tid & 31;
  const int r = lane & 15, h = lane >> 4;
  const int wm = (wave >> 1) * 32;   // wave M offset: 0,32,64,96
  const int wn = (wave & 1) * 64;    // wave N offset: 0,64
  const int bm = blockIdx.y * GEMM_BM;
  const int bn = blockIdx.x * GEMM_BN;

  const int c0row = tid >> 1, c0col = (tid & 1) * 16;   // prefetch address helper

  v8f acc[2][4];
  #pragma unroll
  for (int i = 0; i < 2; ++i)
    #pragma unroll
    for (int j = 0; j < 4; ++j)
      #pragma unroll
      for (int q = 0; q < 8; ++q) acc[i][j][q] = 0.f;

  auto stage = [&](int buf, int k0) {
    #pragma unroll
    for (int t = 0; t < 2; ++t) {
      const int c   = tid + t * 256;     // 0..511 chunks of 8 halfs (16B)
      const int row = c >> 2, col = (c & 3) * 8;
      copy16_gl_to_lds(A + (size_t)(bm + row) * K + k0 + col, &sA[buf][row * LDSK + col]);
      copy16_gl_to_lds(B + (size_t)(bn + row) * K + k0 + col, &sB[buf][row * LDSK + col]);
    }
    if (k0 + GEMM_BK < K) {  // warm L2 for the stage after this one
      __builtin_prefetch(A + (size_t)(bm + c0row) * K + k0 + GEMM_BK + c0col, 0, 1);
      __builtin_prefetch(B + (size_t)(bn + c0row) * K + k0 + GEMM_BK + c0col, 0, 1);
    }
  };

  stage(0, 0);
  wait_async_copies();
  __syncthreads();

  int cur = 0;
  for (int k0 = 0; k0 < K; k0 += GEMM_BK) {
    const int nxt = cur ^ 1;
    if (k0 + GEMM_BK < K) stage(nxt, k0 + GEMM_BK);

    // A fragment (ISA 7.12.2, 16-bit A 16x32): lane L holds row M=L%16,
    // halfs [8h..8h+7] and [16+8h..16+8h+7].  B fragment (32x16): lane L holds
    // column N=L%16, halfs [8h..8h+7] and [16+8h..16+8h+7] (same k order as A).
    v16h afrag[2], bfrag[4];
    #pragma unroll
    for (int i = 0; i < 2; ++i) {
      const _Float16* ap = &sA[cur][(wm + i * 16 + r) * LDSK];
      #pragma unroll
      for (int q = 0; q < 8; ++q) {
        afrag[i][q]     = ap[h * 8 + q];
        afrag[i][8 + q] = ap[16 + h * 8 + q];
      }
    }
    #pragma unroll
    for (int j = 0; j < 4; ++j) {
      const _Float16* bp = &sB[cur][(wn + j * 16 + r) * LDSK];
      #pragma unroll
      for (int q = 0; q < 8; ++q) {
        bfrag[j][q]     = bp[h * 8 + q];
        bfrag[j][8 + q] = bp[16 + h * 8 + q];
      }
    }
    #pragma unroll
    for (int i = 0; i < 2; ++i)
      #pragma unroll
      for (int j = 0; j < 4; ++j)
        acc[i][j] = __builtin_amdgcn_wmma_f32_16x16x32_f16(
            false, afrag[i], false, bfrag[j], (short)0, acc[i][j], false, false);

    if (k0 + GEMM_BK < K) wait_async_copies();
    __syncthreads();
    cur = nxt;
  }

  // C layout: element (v, lane) -> row 8h+v, col r
  #pragma unroll
  for (int i = 0; i < 2; ++i)
    #pragma unroll
    for (int j = 0; j < 4; ++j) {
      const int col = bn + wn + j * 16 + r;
      #pragma unroll
      for (int v = 0; v < 8; ++v) {
        const int row = bm + wm + i * 16 + h * 8 + v;
        C[(size_t)row * N + col] = acc[i][j][v];
      }
    }
}

// ---------------------------------------------------------------------------
__global__ void cluster_accum(const float* __restrict__ rna, const int* __restrict__ label,
                              float* cnts, float* s, float* ss) {
  int i = blockIdx.x * blockDim.x + threadIdx.x;
  if (i >= NN) return;
  int c = label[i];
  atomicAdd(&cnts[c], 1.0f);
  for (int d = 0; d < DD; ++d) {
    float v = rna[(size_t)i * DD + d];
    atomicAdd(&s[c * DD + d], v);
    atomicAdd(&ss[c * DD + d], v * v);
  }
}

__global__ void cluster_finalize(const float* cnts, const float* s, const float* ss,
                                 float* mean_c, float* SC) {
  __shared__ float stdsum[CC];
  const int tid = threadIdx.x;
  if (tid < CC) stdsum[tid] = 0.f;
  __syncthreads();
  for (int idx = tid; idx < CC * DD; idx += 256) {
    int c = idx >> 6;
    float n = cnts[c];
    float m = s[idx] / n;
    mean_c[idx] = m;
    float var = (ss[idx] - n * m * m) / fmaxf(n - 1.f, 1.f);
    atomicAdd(&stdsum[c], sqrtf(fmaxf(var, 0.f)) * (1.f / 64.f));
  }
  __syncthreads();
  if (tid == 0) {
    float csl = 0.f, kmin = 1e30f, raren = 0.f;
    for (int c = 0; c < CC; ++c) {
      float n = cnts[c];
      if (n > 1.f) csl += stdsum[c];
      kmin = fminf(kmin, n);
      if (n < RARE_THR) raren += n;
    }
    SC[SC_CSL]  = csl / (float)CC;
    SC[SC_KK]   = kmin;
    SC[SC_K2]   = fmaxf(2.f, kmin);
    SC[SC_RARE] = raren / (float)NN;
  }
}

// per-column mean/std (ddof=1) + global sum|x|
__global__ __launch_bounds__(256)
void colstats(const float* __restrict__ X, float* colmean, float* sumstd, float* sumabs) {
  const int d = blockIdx.x;
  const int tid = threadIdx.x;
  __shared__ float r0[256], r1[256], r2[256];
  float a = 0.f, b = 0.f, c = 0.f;
  for (int j = tid; j < NN; j += 256) {
    float v = X[(size_t)j * DD + d];
    a += v; b += v * v; c += fabsf(v);
  }
  r0[tid] = a; r1[tid] = b; r2[tid] = c; __syncthreads();
  for (int o = 128; o > 0; o >>= 1) {
    if (tid < o) { r0[tid] += r0[tid + o]; r1[tid] += r1[tid + o]; r2[tid] += r2[tid + o]; }
    __syncthreads();
  }
  if (tid == 0) {
    float m = r0[0] / (float)NN;
    float var = (r1[0] - (float)NN * m * m) / (float)(NN - 1);
    colmean[d] = m;
    atomicAdd(sumstd, sqrtf(fmaxf(var, 0.f)));
    atomicAdd(sumabs, r2[0]);
  }
}

// |cov[a][b]| for a!=b, accumulate positive entries + count
__global__ __launch_bounds__(256)
void cov_offdiag(const float* __restrict__ X, const float* __restrict__ colmean,
                 float* sumAbs, float* cntPos) {
  const int a = blockIdx.x, b = blockIdx.y;
  const int tid = threadIdx.x;
  __shared__ float red[256];
  const float ma = colmean[a], mb = colmean[b];
  float s = 0.f;
  for (int j = tid; j < NN; j += 256)
    s += (X[(size_t)j * DD + a] - ma) * (X[(size_t)j * DD + b] - mb);
  red[tid] = s; __syncthreads();
  for (int o = 128; o > 0; o >>= 1) { if (tid < o) red[tid] += red[tid + o]; __syncthreads(); }
  if (tid == 0 && a != b) {
    float cov = red[0] / (float)(NN - 1);
    float av = fabsf(cov);
    if (av > 0.f) { atomicAdd(sumAbs, av); atomicAdd(cntPos, 1.f); }
  }
}

// pair_sums[ca][cb] += L1(x_i, x_j) over all (i,j); 32x32 pair tiles
__global__ __launch_bounds__(256)
void l1_pairsums(const float* __restrict__ X, const int* __restrict__ label,
                 float* __restrict__ pair_sums) {
  __shared__ float Ai[32 * 64];
  __shared__ float Aj[32 * 64];
  __shared__ int   Li[32], Lj[32];
  __shared__ float accC[CC * CC];
  const int tid = threadIdx.x;
  const int bi = blockIdx.x * 32, bj = blockIdx.y * 32;
  for (int p = tid; p < CC * CC; p += 256) accC[p] = 0.f;
  for (int p = tid; p < 32 * 64; p += 256) {
    int row = p >> 6, col = p & 63;
    Ai[p] = X[(size_t)(bi + row) * DD + col];
    Aj[p] = X[(size_t)(bj + row) * DD + col];
  }
  if (tid < 32) { Li[tid] = label[bi + tid]; Lj[tid] = label[bj + tid]; }
  __syncthreads();
  const int il = tid >> 3, jb = (tid & 7) * 4;
  const int ci = Li[il];
  const float* xi = &Ai[il * 64];
  for (int jj = 0; jj < 4; ++jj) {
    const int jl = jb + jj;
    const float* xj = &Aj[jl * 64];
    float dsum = 0.f;
    for (int d = 0; d < 64; ++d) dsum += fabsf(xi[d] - xj[d]);
    atomicAdd(&accC[ci * CC + Lj[jl]], dsum);
  }
  __syncthreads();
  for (int p = tid; p < CC * CC; p += 256) atomicAdd(&pair_sums[p], accC[p]);
}

// per-row: kk-th largest threshold via binary search on count; adjacency byte mask
__global__ __launch_bounds__(256)
void adjacency_kernel(const float* __restrict__ pc, unsigned char* __restrict__ adj,
                      const float* SC) {
  const int i = blockIdx.x;
  const int tid = threadIdx.x;
  const int kk = (int)SC[SC_KK];
  const float* row = pc + (size_t)i * NN;
  __shared__ int red[256];
  int cnt = 0;
  for (int j = tid; j < NN; j += 256) if (j != i && row[j] >= 0.8f) cnt++;
  red[tid] = cnt; __syncthreads();
  for (int o = 128; o > 0; o >>= 1) { if (tid < o) red[tid] += red[tid + o]; __syncthreads(); }
  const int c08 = red[0]; __syncthreads();

  float lo = 0.8f, hi = 1.0f;
  if (c08 > kk) {
    for (int it = 0; it < 24; ++it) {
      const float mid = 0.5f * (lo + hi);
      int cm = 0;
      for (int j = tid; j < NN; j += 256) if (j != i && row[j] >= mid) cm++;
      red[tid] = cm; __syncthreads();
      for (int o = 128; o > 0; o >>= 1) { if (tid < o) red[tid] += red[tid + o]; __syncthreads(); }
      const int total = red[0]; __syncthreads();
      if (total >= kk) lo = mid; else hi = mid;
    }
  }
  const float thr = fmaxf(lo, 0.8f);
  for (int j = tid; j < NN; j += 256)
    adj[(size_t)i * NN + j] = (unsigned char)((j != i && row[j] >= thr) ? 1 : 0);
}

__global__ __launch_bounds__(256)
void row_argmax(const float* __restrict__ M, int* bidx, float* bval) {
  const int i = blockIdx.x;
  const int tid = threadIdx.x;
  __shared__ float rv[256]; __shared__ int ri[256];
  float bv = -1e30f; int bx = 0;
  const float* row = M + (size_t)i * NN;
  for (int j = tid; j < NN; j += 256) {
    float v = row[j];
    if (v > bv) { bv = v; bx = j; }
  }
  rv[tid] = bv; ri[tid] = bx; __syncthreads();
  for (int o = 128; o > 0; o >>= 1) {
    if (tid < o) {
      if (rv[tid + o] > rv[tid] || (rv[tid + o] == rv[tid] && ri[tid + o] < ri[tid])) {
        rv[tid] = rv[tid + o]; ri[tid] = ri[tid + o];
      }
    }
    __syncthreads();
  }
  if (tid == 0) { bidx[i] = ri[0]; bval[i] = rv[0]; }
}

// per-column k2-th largest of M[:,c] via binary search
__global__ __launch_bounds__(256)
void col_kth_thresh(const float* __restrict__ M, const float* SC, float* t2) {
  const int c = blockIdx.x;
  const int tid = threadIdx.x;
  const int k2 = (int)SC[SC_K2];
  __shared__ int red[256];
  float lo = -1.001f, hi = 1.001f;
  for (int it = 0; it < 28; ++it) {
    const float mid = 0.5f * (lo + hi);
    int cnt = 0;
    for (int j = tid; j < NN; j += 256) if (M[(size_t)j * NN + c] >= mid) cnt++;
    red[tid] = cnt; __syncthreads();
    for (int o = 128; o > 0; o >>= 1) { if (tid < o) red[tid] += red[tid + o]; __syncthreads(); }
    const int total = red[0]; __syncthreads();
    if (total >= k2) lo = mid; else hi = mid;
  }
  if (tid == 0) t2[c] = lo;
}

__global__ void center_arg_kernel(const float* __restrict__ atac,
                                  const float* __restrict__ mean_c, int* carg) {
  const int j = blockIdx.x * blockDim.x + threadIdx.x;
  if (j >= NN) return;
  float best = -1e30f; int ba = 0;
  for (int c = 0; c < CC; ++c) {
    float dot = 0.f, nn = 0.f;
    for (int d = 0; d < DD; ++d) {
      float m = mean_c[c * DD + d];
      dot += atac[(size_t)j * DD + d] * m;
      nn += m * m;
    }
    float v = dot * rsqrtf(nn + 1e-30f);
    if (v > best) { best = v; ba = c; }
  }
  carg[j] = ba;
}

__global__ void mnn_match(const float* __restrict__ ar, const float* __restrict__ t2,
                          const int* __restrict__ bidx, const float* __restrict__ bval,
                          const int* __restrict__ label, const float* __restrict__ mean_c,
                          const float* __restrict__ cnts, const float* __restrict__ atac,
                          unsigned char* matched, int* type_i, float* Lmnn, float* cnt1) {
  const int i = blockIdx.x * blockDim.x + threadIdx.x;
  if (i >= NN) return;
  const int b = bidx[i];
  const int ty = label[b];
  type_i[i] = ty;
  const int ok = (ar[(size_t)i * NN + b] >= t2[b]) && (bval[i] > 0.5f);
  matched[i] = (unsigned char)ok;
  if (ok) {
    float a = 0.f;
    for (int d = 0; d < DD; ++d) a += fabsf(atac[(size_t)i * DD + d] - mean_c[ty * DD + d]);
    a *= (1.f / 64.f);
    const float rare = (cnts[ty] < RARE_THR) ? 0.25f : 0.f;
    atomicAdd(Lmnn, (1.f + rare) * a);
    atomicAdd(cnt1, 1.f);
  }
}

__global__ __launch_bounds__(256)
void near_loss_kernel(const float* __restrict__ atac, const unsigned char* __restrict__ adj,
                      float* nearAcc) {
  const int i = blockIdx.x;
  const int tid = threadIdx.x;
  __shared__ float w[NN];
  __shared__ float sd[DD];
  __shared__ float nw_s;
  for (int j = tid; j < NN; j += 256) w[j] = adj[(size_t)i * NN + j] ? 1.f : 0.f;
  __syncthreads();
  if (tid == 0) w[i] += 1.f;
  __syncthreads();
  if (tid < DD) {
    float s = 0.f, s2 = 0.f, nw = 0.f;
    for (int j = 0; j < NN; ++j) {
      if (w[j] > 0.f) {
        float v = atac[(size_t)j * DD + tid];
        s += v; s2 += v * v; nw += 1.f;
      }
    }
    float mw = s / nw;
    float vw = (s2 - nw * mw * mw) / fmaxf(nw - 1.f, 1.f);
    sd[tid] = sqrtf(fmaxf(vw, 0.f));
    if (tid == 0) nw_s = nw;
  }
  __syncthreads();
  if (tid == 0) {
    float sii = 0.f;
    for (int d = 0; d < DD; ++d) sii += sd[d];
    sii *= (1.f / 64.f);
    if (nw_s > 1.f) atomicAdd(nearAcc, sii);
  }
}

__global__ __launch_bounds__(256)
void mnn_near(const float* __restrict__ atac, const unsigned char* __restrict__ adj,
              const unsigned char* __restrict__ matched, const int* __restrict__ type_i,
              const int* __restrict__ carg, const float* __restrict__ mean_c,
              const float* __restrict__ cnts, float* Lnear, float* cnt2) {
  const int i = blockIdx.x;
  if (!matched[i]) return;
  const int tid = threadIdx.x;
  const int ty = type_i[i];
  __shared__ float redf[256]; __shared__ int redi[256];
  float acc = 0.f; int cnt = 0;
  const unsigned char* arow = adj + (size_t)i * NN;
  for (int j = tid; j < NN; j += 256) {
    if (arow[j] && !matched[j] && carg[j] == ty) {
      float a = 0.f;
      for (int d = 0; d < DD; ++d) a += fabsf(atac[(size_t)j * DD + d] - mean_c[ty * DD + d]);
      acc += a * (1.f / 64.f);
      cnt++;
    }
  }
  redf[tid] = acc; redi[tid] = cnt; __syncthreads();
  for (int o = 128; o > 0; o >>= 1) {
    if (tid < o) { redf[tid] += redf[tid + o]; redi[tid] += redi[tid + o]; }
    __syncthreads();
  }
  if (tid == 0) {
    const float rare = (cnts[ty] < RARE_THR) ? 0.25f : 0.f;
    atomicAdd(Lnear, 0.8f * (1.f + rare) * redf[0]);
    atomicAdd(cnt2, (float)redi[0]);
  }
}

__global__ void finalize_kernel(const float* SC, const float* pair_sums,
                                const float* cnts, float* out) {
  float dmm = 0.f;
  for (int a = 0; a < CC; ++a)
    for (int b = 0; b < CC; ++b) {
      if (a == b) continue;
      dmm += pair_sums[a * CC + b] / (cnts[a] * cnts[b] * (float)DD);
    }
  dmm /= (float)(CC * CC);
  const float w = 2.f * (SC[SC_RARE] + 0.01f);           // r10 = round(C/10) = 2
  const float meanstd_r = SC[SC_SUMSTD_R] / (float)DD;
  const float meanabs_r = SC[SC_SUMABS_R] / (float)(NN * DD);
  const float off_r = SC[SC_COVS_R] / fmaxf(SC[SC_COVC_R], 1.f);
  const float rna_red = -w * dmm + (1.f - w) / meanstd_r + SC[SC_CSL]
                      + 2.f * off_r + 0.1f * meanabs_r;  // 2/C = 0.1
  const float meanstd_a = SC[SC_SUMSTD_A] / (float)DD;
  const float meanabs_a = SC[SC_SUMABS_A] / (float)(NN * DD);
  const float off_a = SC[SC_COVS_A] / fmaxf(SC[SC_COVC_A], 1.f);
  const float atac_red = (1.f - w) / meanstd_a + 2.f * off_a + 0.1f * meanabs_a;
  const float near = SC[SC_NEAR] / (float)NN;
  const float mnn = SC[SC_LMNN] / fmaxf(SC[SC_CNT1], 1.f)
                  + SC[SC_LNEAR] / fmaxf(SC[SC_CNT2], 1.f);
  out[0] = rna_red + atac_red + near + mnn;
}

// ---------------------------------------------------------------------------
extern "C" void kernel_launch(void* const* d_in, const int* in_sizes, int n_in,
                              void* d_out, int out_size, void* d_ws, size_t ws_size,
                              hipStream_t stream) {
  (void)in_sizes; (void)n_in; (void)out_size; (void)ws_size;
  const float* atac  = (const float*)d_in[0];
  const float* rna   = (const float*)d_in[1];
  const float* peak  = (const float*)d_in[2];
  const int*   label = (const int*)d_in[3];
  float* out = (float*)d_out;

  char* ws = (char*)d_ws;
  _Float16* Pn    = (_Float16*)(ws + PN_OFF);
  float*    gm    = (float*)(ws + GM_OFF);          // pc, then reused as ar
  unsigned char* adj = (unsigned char*)(ws + ADJ_OFF);
  _Float16* atacn = (_Float16*)(ws + ATACN_OFF);
  _Float16* rnan  = (_Float16*)(ws + RNAN_OFF);
  float* F = (float*)(ws + FREG_OFF);
  float* cnts  = F + FI_CNTS;
  float* s_acc = F + FI_S;
  float* ss_acc= F + FI_SS;
  float* psums = F + FI_PS;
  float* SC    = F + FI_SC;
  float* meanc = F + FI_MEANC;
  float* cmr   = F + FI_CMR;
  float* cma   = F + FI_CMA;
  float* t2    = F + FI_T2;
  float* bsim  = F + FI_BSIM;
  int* iwork   = (int*)(ws + INT_OFF);
  int* bidx    = iwork;
  int* typei   = iwork + NN;
  int* carg    = iwork + 2 * NN;
  unsigned char* matched = (unsigned char*)(ws + MATCH_OFF);

  // 1. zero accumulators
  zero_f32_kernel<<<(ZERO_N + 255) / 256, 256, 0, stream>>>(F, ZERO_N);
  // 2. normalize rows -> f16
  rownorm_to_f16<<<NN, 256, 0, stream>>>(peak, Pn, PP, PPAD);
  rownorm_to_f16<<<NN, 256, 0, stream>>>(atac, atacn, DD, DD);
  rownorm_to_f16<<<NN, 256, 0, stream>>>(rna,  rnan,  DD, DD);
  // 3. pc = Pn * Pn^T  (WMMA, the dominant 42 GFLOP GEMM)
  gemm_nt_f16<<<dim3(NN / GEMM_BN, NN / GEMM_BM), 256, 0, stream>>>(Pn, Pn, gm, NN, NN, PPAD);
  // 4. cluster stats
  cluster_accum<<<(NN + 255) / 256, 256, 0, stream>>>(rna, label, cnts, s_acc, ss_acc);
  cluster_finalize<<<1, 256, 0, stream>>>(cnts, s_acc, ss_acc, meanc, SC);
  // 5. column stats + cov offdiag + L1 pair sums
  colstats<<<DD, 256, 0, stream>>>(rna,  cmr, &SC[SC_SUMSTD_R], &SC[SC_SUMABS_R]);
  colstats<<<DD, 256, 0, stream>>>(atac, cma, &SC[SC_SUMSTD_A], &SC[SC_SUMABS_A]);
  cov_offdiag<<<dim3(DD, DD), 256, 0, stream>>>(rna,  cmr, &SC[SC_COVS_R], &SC[SC_COVC_R]);
  cov_offdiag<<<dim3(DD, DD), 256, 0, stream>>>(atac, cma, &SC[SC_COVS_A], &SC[SC_COVC_A]);
  l1_pairsums<<<dim3(NN / 32, NN / 32), 256, 0, stream>>>(rna, label, psums);
  // 6. knn adjacency from pc (kk-th largest per row, >= 0.8)
  adjacency_kernel<<<NN, 256, 0, stream>>>(gm, adj, SC);
  // 7. ar = cos(atac, rna) reusing gm buffer (WMMA)
  gemm_nt_f16<<<dim3(NN / GEMM_BN, NN / GEMM_BM), 256, 0, stream>>>(atacn, rnan, gm, NN, NN, DD);
  // 8. mutual NN machinery
  row_argmax<<<NN, 256, 0, stream>>>(gm, bidx, bsim);
  col_kth_thresh<<<NN, 256, 0, stream>>>(gm, SC, t2);
  center_arg_kernel<<<(NN + 255) / 256, 256, 0, stream>>>(atac, meanc, carg);
  mnn_match<<<(NN + 255) / 256, 256, 0, stream>>>(gm, t2, bidx, bsim, label, meanc, cnts,
                                                  atac, matched, typei,
                                                  &SC[SC_LMNN], &SC[SC_CNT1]);
  near_loss_kernel<<<NN, 256, 0, stream>>>(atac, adj, &SC[SC_NEAR]);
  mnn_near<<<NN, 256, 0, stream>>>(atac, adj, matched, typei, carg, meanc, cnts,
                                   &SC[SC_LNEAR], &SC[SC_CNT2]);
  // 9. combine
  finalize_kernel<<<1, 1, 0, stream>>>(SC, psums, cnts, out);
}
